// HetGNN_37366215475388
// MI455X (gfx1250) — matrix operand order repaired and checked by reading, not compiled, not run on hardware
//
#include <hip/hip_runtime.h>
#include <hip/hip_bf16.h>
#include <math.h>

typedef __attribute__((ext_vector_type(16))) _Float16 v16h;
typedef __attribute__((ext_vector_type(8)))  float    v8f;

#define FDIM 128   // H*C = 2*64

// ---------------------------------------------------------------------------
// float atomic max via signed/unsigned integer ordering trick
// ---------------------------------------------------------------------------
__device__ __forceinline__ void atomic_max_float(float* addr, float val) {
  if (val >= 0.0f) atomicMax((int*)addr, __float_as_int(val));
  else             atomicMin((unsigned int*)addr, __float_as_uint(val));
}

// ---------------------------------------------------------------------------
// WMMA GEMM:  Y[M,N] = act( A[M,K] @ W[K,N] + bias )
//   K multiple of 32, N = NTILES*16, M multiple of 16.
//   Block = 256 threads = 8 waves. Each wave owns a 16-row strip and all
//   NTILES column tiles. W is staged transposed (f16) in LDS once per block.
//   MODE: 0 = none, 2 = bias+relu, 3 = bias+relu+sigmoid
// ---------------------------------------------------------------------------
template<int K, int NTILES, int MODE>
__global__ __launch_bounds__(256) void wmma_gemm_kernel(
    const float* __restrict__ A, const float* __restrict__ W,
    const float* __restrict__ bias, float* __restrict__ Y, int M)
{
  constexpr int N = NTILES * 16;
  constexpr int KSTEPS = K / 32;
  __shared__ __attribute__((aligned(32))) _Float16 ldsW[(size_t)N * K]; // [n][k]

  const int tid = threadIdx.x;
  // cooperative load + transpose + f32->f16 convert of W
  for (int i = tid; i < K * N; i += 256) {
    int k = i / N, n = i % N;
    ldsW[n * K + k] = (_Float16)W[i];
  }
  __syncthreads();

  const int wave = tid >> 5;
  const int lane = tid & 31;
  const int mtile = (blockIdx.x * 8 + wave) * 16;
  if (mtile >= M) return;                 // wave-uniform: EXEC stays full

  const int mrow = mtile + (lane & 15);
  const int kbA  = (lane >> 4) * 16;      // contiguous k order: lane half h holds k 16h..16h+15
  const int kbB  = (lane >> 4) * 16;      // same contiguous k order as A (k 16h..16h+15)
  const int ncol = lane & 15;

  const v8f vzero = {0.f,0.f,0.f,0.f,0.f,0.f,0.f,0.f};
  v8f acc[NTILES];
#pragma unroll
  for (int t = 0; t < NTILES; ++t) acc[t] = vzero;

  const float* arow = A + (size_t)mrow * K;
  for (int ks = 0; ks < KSTEPS; ++ks) {
    const int k0 = ks * 32;
    v16h afrag;
#pragma unroll
    for (int j = 0; j < 8; ++j) {
      afrag[j]     = (_Float16)arow[k0 + kbA + j];
      afrag[8 + j] = (_Float16)arow[k0 + kbA + 8 + j];
    }
#pragma unroll
    for (int t = 0; t < NTILES; ++t) {
      const v16h bfrag = *(const v16h*)&ldsW[(size_t)(t * 16 + ncol) * K + k0 + kbB];
      acc[t] = __builtin_amdgcn_wmma_f32_16x16x32_f16(
          false, afrag, false, bfrag, (short)0, acc[t], false, false);
    }
  }

  // epilogue: C/D layout -> VGPR r holds M=r (lanes 0-15) / M=r+8 (lanes 16-31)
  const int mbase = mtile + ((lane >> 4) * 8);
#pragma unroll
  for (int t = 0; t < NTILES; ++t) {
    const int n = t * 16 + ncol;
    float b = 0.0f;
    if (MODE != 0) b = bias[n];
#pragma unroll
    for (int r = 0; r < 8; ++r) {
      float v = acc[t][r] + b;
      if (MODE >= 2) v = fmaxf(v, 0.0f);
      if (MODE == 3) v = 1.0f / (1.0f + expf(-v));
      Y[(size_t)(mbase + r) * N + n] = v;
    }
  }
}

// ---------------------------------------------------------------------------
// fill
// ---------------------------------------------------------------------------
__global__ void fill_kernel(float* __restrict__ p, float v, int n) {
  int i = blockIdx.x * 256 + threadIdx.x;
  if (i < n) p[i] = v;
}

// ---------------------------------------------------------------------------
// edge scores: one wave per edge; lanes 0-15 head 0, lanes 16-31 head 1,
// 4 features per lane; shuffle-reduce inside each half-wave; atomic segment max
// ---------------------------------------------------------------------------
__global__ __launch_bounds__(256) void edge_score_kernel(
    const int* __restrict__ esrc, const int* __restrict__ etgt,
    const float* __restrict__ xl, const float* __restrict__ xr,
    const float* __restrict__ att,          // [2*64] flat
    float* __restrict__ score, float* __restrict__ smax, int E)
{
  const int lane = threadIdx.x & 31;
  const int wid  = (blockIdx.x * 256 + threadIdx.x) >> 5;
  const int nw   = (gridDim.x * 256) >> 5;
  for (int e = wid; e < E; e += nw) {
    const int s = esrc[e], t = etgt[e];
    const int en = e + nw;
    if (en < E) {   // gfx1250 global_prefetch for next edge's rows
      __builtin_prefetch(&xl[(size_t)esrc[en] * FDIM], 0, 1);
      __builtin_prefetch(&xr[(size_t)etgt[en] * FDIM], 0, 1);
    }
    const float* pl = xl + (size_t)s * FDIM;
    const float* pr = xr + (size_t)t * FDIM;
    float partial = 0.0f;
#pragma unroll
    for (int j = 0; j < 4; ++j) {
      const int f = lane * 4 + j;
      float z = pl[f] + pr[f];
      z = (z > 0.0f) ? z : 0.2f * z;       // LeakyReLU(0.2)
      partial += z * att[f];
    }
    partial += __shfl_xor(partial, 1, 32);
    partial += __shfl_xor(partial, 2, 32);
    partial += __shfl_xor(partial, 4, 32);
    partial += __shfl_xor(partial, 8, 32);
    if ((lane & 15) == 0) {
      const int h = lane >> 4;
      score[(size_t)e * 2 + h] = partial;
      atomic_max_float(&smax[t * 2 + h], partial);
    }
  }
}

// ---------------------------------------------------------------------------
// exp(score - smax) and segment denominator
// ---------------------------------------------------------------------------
__global__ void edge_exp_kernel(const int* __restrict__ etgt,
                                float* __restrict__ score,
                                const float* __restrict__ smax,
                                float* __restrict__ den, int E)
{
  int i = blockIdx.x * 256 + threadIdx.x;        // over E*2
  if (i >= E * 2) return;
  const int e = i >> 1, h = i & 1;
  const int t = etgt[e];
  float m = smax[t * 2 + h];
  if (!isfinite(m)) m = 0.0f;
  const float ex = expf(score[i] - m);
  score[i] = ex;
  atomicAdd(&den[t * 2 + h], ex);
}

// ---------------------------------------------------------------------------
// weighted scatter aggregation: out[tgt] += alpha * xl[src]
// ---------------------------------------------------------------------------
__global__ __launch_bounds__(256) void edge_aggr_kernel(
    const int* __restrict__ esrc, const int* __restrict__ etgt,
    const float* __restrict__ xl, const float* __restrict__ score,
    const float* __restrict__ den, float* __restrict__ out, int E)
{
  const int lane = threadIdx.x & 31;
  const int wid  = (blockIdx.x * 256 + threadIdx.x) >> 5;
  const int nw   = (gridDim.x * 256) >> 5;
  for (int e = wid; e < E; e += nw) {
    const int s = esrc[e], t = etgt[e];
    const float a0 = score[(size_t)e * 2 + 0] / fmaxf(den[t * 2 + 0], 1e-16f);
    const float a1 = score[(size_t)e * 2 + 1] / fmaxf(den[t * 2 + 1], 1e-16f);
    const float* pl = xl + (size_t)s * FDIM;
    float* po = out + (size_t)t * FDIM;
#pragma unroll
    for (int j = 0; j < 4; ++j) {
      const int f = lane * 4 + j;
      const float alpha = (f < 64) ? a0 : a1;
      atomicAdd(&po[f], alpha * pl[f]);
    }
  }
}

// ---------------------------------------------------------------------------
// h_l finalize: in-place bias add
// ---------------------------------------------------------------------------
__global__ void bias_add_kernel(float* __restrict__ h, const float* __restrict__ b, int n) {
  int i = blockIdx.x * 256 + threadIdx.x;
  if (i < n) h[i] += b[i & (FDIM - 1)];
}

// h_m = 0.5*((h_mm+b_mm) + (h_lm+b_lm)), written into h_mm
__global__ void combine_kernel(float* __restrict__ hmm, const float* __restrict__ hlm,
                               const float* __restrict__ bmm, const float* __restrict__ blm,
                               int n) {
  int i = blockIdx.x * 256 + threadIdx.x;
  if (i < n) {
    const int f = i & (FDIM - 1);
    hmm[i] = 0.5f * ((hmm[i] + bmm[f]) + (hlm[i] + blm[f]));
  }
}

// cat[p, 0:128] = h_l[pairs[p,0]], cat[p, 128:256] = h_m[pairs[p,1]]
__global__ void gather_concat_kernel(const float* __restrict__ h_l,
                                     const float* __restrict__ h_m,
                                     const int* __restrict__ pairs,
                                     float* __restrict__ cat, int P)
{
  int i = blockIdx.x * 256 + threadIdx.x;        // over P*256
  if (i >= P * 256) return;
  const int p = i >> 8, f = i & 255;
  const int idx = (f < FDIM) ? pairs[p * 2 + 0] : pairs[p * 2 + 1];
  const float* src = (f < FDIM) ? h_l : h_m;
  cat[i] = src[(size_t)idx * FDIM + (f & (FDIM - 1))];
}

// fused = g*f1 + (1-g)*f2
__global__ void fuse_kernel(const float* __restrict__ g,
                            const float* __restrict__ h_l,
                            const float* __restrict__ h_m,
                            const int* __restrict__ pairs,
                            float* __restrict__ fused, int P)
{
  int i = blockIdx.x * 256 + threadIdx.x;        // over P*128
  if (i >= P * FDIM) return;
  const int p = i >> 7, f = i & (FDIM - 1);
  const float f1 = h_l[(size_t)pairs[p * 2 + 0] * FDIM + f];
  const float f2 = h_m[(size_t)pairs[p * 2 + 1] * FDIM + f];
  const float gg = g[i];
  fused[i] = gg * f1 + (1.0f - gg) * f2;
}

// out[p] = dot(h2[p,:32], c3_W) + c3_b  — one wave per pair
__global__ __launch_bounds__(256) void final_dot_kernel(
    const float* __restrict__ h2, const float* __restrict__ w,
    const float* __restrict__ b, float* __restrict__ out, int P)
{
  const int lane = threadIdx.x & 31;
  const int p = (blockIdx.x * 256 + threadIdx.x) >> 5;
  if (p >= P) return;
  float v = h2[(size_t)p * 32 + lane] * w[lane];
  v += __shfl_xor(v, 1, 32);
  v += __shfl_xor(v, 2, 32);
  v += __shfl_xor(v, 4, 32);
  v += __shfl_xor(v, 8, 32);
  v += __shfl_xor(v, 16, 32);
  if (lane == 0) out[p] = v + b[0];
}

// ---------------------------------------------------------------------------
// launcher
// ---------------------------------------------------------------------------
extern "C" void kernel_launch(void* const* d_in, const int* in_sizes, int n_in,
                              void* d_out, int out_size, void* d_ws, size_t ws_size,
                              hipStream_t stream)
{
  const float* x_lnc  = (const float*)d_in[0];
  const float* x_mi   = (const float*)d_in[1];
  const float* Wl_ll  = (const float*)d_in[2];
  const float* Wr_ll  = (const float*)d_in[3];
  const float* att_ll = (const float*)d_in[4];
  const float* b_ll   = (const float*)d_in[5];
  const float* Wl_mm  = (const float*)d_in[6];
  const float* Wr_mm  = (const float*)d_in[7];
  const float* att_mm = (const float*)d_in[8];
  const float* b_mm   = (const float*)d_in[9];
  const float* Wl_lm  = (const float*)d_in[10];
  const float* Wr_lm  = (const float*)d_in[11];
  const float* att_lm = (const float*)d_in[12];
  const float* b_lm   = (const float*)d_in[13];
  const float* gate_W = (const float*)d_in[14];
  const float* gate_b = (const float*)d_in[15];
  const float* c1_W   = (const float*)d_in[16];
  const float* c1_b   = (const float*)d_in[17];
  const float* c2_W   = (const float*)d_in[18];
  const float* c2_b   = (const float*)d_in[19];
  const float* c3_W   = (const float*)d_in[20];
  const float* c3_b   = (const float*)d_in[21];
  const int* edge_ll  = (const int*)d_in[22];
  const int* edge_mm  = (const int*)d_in[23];
  const int* edge_lm  = (const int*)d_in[24];
  const int* pairs    = (const int*)d_in[25];

  const int IN = 256;
  const int N  = in_sizes[0] / IN;          // 30000
  const int E  = in_sizes[22] / 2;          // 480000
  const int P  = in_sizes[25] / 2;          // 100000

  // -------- workspace bump allocator --------
  char* p = (char*)d_ws;
  auto alloc_f = [&](size_t n) -> float* {
    float* r = (float*)p;
    p += (n * sizeof(float) + 255) & ~(size_t)255;
    return r;
  };
  // projection + score region (dead after convs) is reused for `cat`
  float* cat   = (float*)d_ws;
  float* xl_ll = alloc_f((size_t)N * FDIM);
  float* xr_ll = alloc_f((size_t)N * FDIM);
  float* xl_mm = alloc_f((size_t)N * FDIM);
  float* xr_mm = alloc_f((size_t)N * FDIM);
  float* xl_lm = alloc_f((size_t)N * FDIM);
  float* xr_lm = alloc_f((size_t)N * FDIM);
  float* sc_ll = alloc_f((size_t)E * 2);
  float* sc_mm = alloc_f((size_t)E * 2);
  float* sc_lm = alloc_f((size_t)E * 2);
  {
    size_t region = (size_t)(p - (char*)d_ws);
    size_t need   = (size_t)P * 256 * sizeof(float);
    if (region < need) p = (char*)d_ws + ((need + 255) & ~(size_t)255);
  }
  float* smax_ll = alloc_f((size_t)N * 2);
  float* smax_mm = alloc_f((size_t)N * 2);
  float* smax_lm = alloc_f((size_t)N * 2);
  float* den_ll  = alloc_f((size_t)N * 2);
  float* den_mm  = alloc_f((size_t)N * 2);
  float* den_lm  = alloc_f((size_t)N * 2);
  float* hacc_ll = alloc_f((size_t)N * FDIM);   // becomes h_l
  float* hacc_mm = alloc_f((size_t)N * FDIM);   // becomes h_m
  float* hacc_lm = alloc_f((size_t)N * FDIM);
  float* g_buf   = alloc_f((size_t)P * FDIM);
  float* fused   = alloc_f((size_t)P * FDIM);
  float* h1      = alloc_f((size_t)P * 64);
  float* h2      = alloc_f((size_t)P * 32);

  const dim3 blk(256);
  const int gN  = (N + 127) / 128;
  const int gP  = (P + 127) / 128;
  const int nNF = N * FDIM;

  // -------- projections (WMMA f16 GEMMs) --------
  wmma_gemm_kernel<256, 8, 0><<<gN, blk, 0, stream>>>(x_lnc, Wl_ll, nullptr, xl_ll, N);
  wmma_gemm_kernel<256, 8, 0><<<gN, blk, 0, stream>>>(x_lnc, Wr_ll, nullptr, xr_ll, N);
  wmma_gemm_kernel<256, 8, 0><<<gN, blk, 0, stream>>>(x_mi,  Wl_mm, nullptr, xl_mm, N);
  wmma_gemm_kernel<256, 8, 0><<<gN, blk, 0, stream>>>(x_mi,  Wr_mm, nullptr, xr_mm, N);
  wmma_gemm_kernel<256, 8, 0><<<gN, blk, 0, stream>>>(x_lnc, Wl_lm, nullptr, xl_lm, N);
  wmma_gemm_kernel<256, 8, 0><<<gN, blk, 0, stream>>>(x_mi,  Wr_lm, nullptr, xr_lm, N);

  // -------- per-conv edge softmax + aggregation --------
  auto run_conv = [&](const int* edge, const float* xl, const float* xr,
                      const float* att, float* sc, float* smax, float* den,
                      float* hacc) {
    const int eb = (E + 7) / 8;                       // one wave per edge
    fill_kernel<<<(2 * N + 255) / 256, blk, 0, stream>>>(smax, -INFINITY, 2 * N);
    fill_kernel<<<(2 * N + 255) / 256, blk, 0, stream>>>(den, 0.0f, 2 * N);
    fill_kernel<<<(nNF + 255) / 256, blk, 0, stream>>>(hacc, 0.0f, nNF);
    edge_score_kernel<<<eb, blk, 0, stream>>>(edge, edge + E, xl, xr, att, sc, smax, E);
    edge_exp_kernel<<<(2 * E + 255) / 256, blk, 0, stream>>>(edge + E, sc, smax, den, E);
    edge_aggr_kernel<<<eb, blk, 0, stream>>>(edge, edge + E, xl, sc, den, hacc, E);
  };
  run_conv(edge_ll, xl_ll, xr_ll, att_ll, sc_ll, smax_ll, den_ll, hacc_ll);
  run_conv(edge_mm, xl_mm, xr_mm, att_mm, sc_mm, smax_mm, den_mm, hacc_mm);
  run_conv(edge_lm, xl_lm, xr_lm, att_lm, sc_lm, smax_lm, den_lm, hacc_lm);

  // -------- finalize node embeddings --------
  bias_add_kernel<<<(nNF + 255) / 256, blk, 0, stream>>>(hacc_ll, b_ll, nNF);
  combine_kernel<<<(nNF + 255) / 256, blk, 0, stream>>>(hacc_mm, hacc_lm, b_mm, b_lm, nNF);

  // -------- pair head --------
  gather_concat_kernel<<<(P * 256 + 255) / 256, blk, 0, stream>>>(hacc_ll, hacc_mm, pairs, cat, P);
  wmma_gemm_kernel<256, 8, 3><<<gP, blk, 0, stream>>>(cat, gate_W, gate_b, g_buf, P);
  fuse_kernel<<<(P * FDIM + 255) / 256, blk, 0, stream>>>(g_buf, hacc_ll, hacc_mm, pairs, fused, P);
  wmma_gemm_kernel<128, 4, 2><<<gP, blk, 0, stream>>>(fused, c1_W, c1_b, h1, P);
  wmma_gemm_kernel<64, 2, 2><<<gP, blk, 0, stream>>>(h1, c2_W, c2_b, h2, P);
  final_dot_kernel<<<(P + 7) / 8, blk, 0, stream>>>(h2, c3_W, c3_b, (float*)d_out, P);
}
